// ResBlock_d3d_12850542149997
// MI455X (gfx1250) — matrix operand order repaired and checked by reading, not compiled, not run on hardware
//
#include <hip/hip_runtime.h>
#include <hip/hip_bf16.h>

typedef __attribute__((ext_vector_type(16))) _Float16 v16h;
typedef __attribute__((ext_vector_type(8)))  _Float16 v8h;
typedef __attribute__((ext_vector_type(8)))  float    v8f;
typedef __attribute__((ext_vector_type(4)))  int      v4i;

#define NF    64
#define TAPS  27
#define OFFC  54
#define TT    7
#define HH    96
#define WW    96
#define HW    (HH*WW)
#define THW   (TT*HH*WW)      // 64512
#define NPOS  THW
#define NTILE 64
#define NTILES (NPOS/NTILE)   // 1008 exact
#define KSTEPS 2
#define MBLK   4
#define TAPFRAG (KSTEPS*MBLK*512)           // 4096 halves = 8KB per tap
#define FRAGTOTAL (TAPS*TAPFRAG)            // 110592 halves per tensor

#define AS1 __attribute__((address_space(1)))
#define AS3 __attribute__((address_space(3)))

#if __has_builtin(__builtin_amdgcn_global_load_async_to_lds_b128) && \
    __has_builtin(__builtin_amdgcn_s_wait_asynccnt)
#define USE_ASYNC_LDS 1
#define WAIT_ASYNC(n) __builtin_amdgcn_s_wait_asynccnt(n)
#else
#define USE_ASYNC_LDS 0
#define WAIT_ASYNC(n) ((void)0)
#endif

// Copy one tap's A-fragments (8KB) global->LDS; 128 threads x 4 chunks of 16B.
__device__ __forceinline__ void copy_tap_A(const _Float16* __restrict__ src,
                                           _Float16* __restrict__ dst, int tid) {
#pragma unroll
  for (int j = 0; j < 4; ++j) {
    int e = (tid + j * 128) * 8;   // halves
#if USE_ASYNC_LDS
    __builtin_amdgcn_global_load_async_to_lds_b128(
        (AS1 v4i*)(src + e), (AS3 v4i*)(dst + e), 0, 0);
#else
    *(v8h*)(dst + e) = *(const v8h*)(src + e);
#endif
  }
}

// ---------------------------------------------------------------------------
__global__ void cvt_f32_to_f16_kernel(const float* __restrict__ src,
                                      _Float16* __restrict__ dst, int n) {
  int i = blockIdx.x * blockDim.x + threadIdx.x;
  if (i < n) dst[i] = (_Float16)src[i];
}

// ---------------------------------------------------------------------------
// Pack OIDHW fp32 weights (orows valid, zero-pad to 64) into f16 WMMA
// A-fragments, 16x16x32, contiguous k order (B uses same): element i, lane l (g=l/16):
// m = l%16, k = i+16g. Lane reads 16 contiguous halves.
// ---------------------------------------------------------------------------
__global__ void pack_wfrag_kernel(const float* __restrict__ wsrc, int orows,
                                  _Float16* __restrict__ dst) {
  int idx = blockIdx.x * blockDim.x + threadIdx.x;
  if (idx >= FRAGTOTAL) return;
  int i    = idx & 15;
  int lane = (idx >> 4) & 31;
  int mb   = (idx >> 9) & 3;
  int ks   = (idx >> 11) & 1;
  int tap  = idx >> 12;
  int g    = lane >> 4;
  int m    = mb * 16 + (lane & 15);
  int kl   = i + 16 * g;
  int c    = ks * 32 + kl;
  float v  = 0.f;
  if (m < orows) v = wsrc[(m * NF + c) * TAPS + tap];
  dst[idx] = (_Float16)v;
}

// ---------------------------------------------------------------------------
// Offset conv: dense 3x3x3 conv, 64 -> 54 ch, implicit GEMM with WMMA.
// Input x is pre-converted f16. Double-buffered async A-fragments in LDS.
// ---------------------------------------------------------------------------
__global__ __launch_bounds__(128)
void offset_conv_kernel(const _Float16* __restrict__ x,
                        const _Float16* __restrict__ wfrag,
                        const float* __restrict__ bias,
                        float* __restrict__ out) {
  __shared__ alignas(32) _Float16 lds_s[NTILE * NF];     // B: [n][c]
  __shared__ alignas(16) _Float16 lds_a[2][TAPFRAG];     // A double buffer
  __shared__ int s_addr[NTILE];
  __shared__ int s_val[NTILE];

  const int tid  = threadIdx.x;
  const int lane = tid & 31;
  const int wid  = tid >> 5;
  const int g    = lane >> 4;
  const int nl   = lane & 15;
  const int base = blockIdx.x * NTILE;

  const v8f vzero = {0.f, 0.f, 0.f, 0.f, 0.f, 0.f, 0.f, 0.f};
  v8f acc[4];
  acc[0] = vzero; acc[1] = vzero; acc[2] = vzero; acc[3] = vzero;

  copy_tap_A(wfrag, lds_a[0], tid);   // prologue: tap 0 A-fragments

  for (int tap = 0; tap < TAPS; ++tap) {
    const int dt = tap / 9 - 1;
    const int dh = (tap / 3) % 3 - 1;
    const int dw = tap % 3 - 1;

    if (tid < NTILE) {
      int pos = base + tid;
      int t = pos / HW;  int r = pos - t * HW;
      int h = r / WW;    int w = r - h * WW;
      int ts = t + dt, hs = h + dh, wsx = w + dw;
      int valid = (ts >= 0) & (ts < TT) & (hs >= 0) & (hs < HH) &
                  (wsx >= 0) & (wsx < WW);
      s_addr[tid] = valid ? (ts * HW + hs * WW + wsx) : 0;
      s_val[tid]  = valid;
    }
    if (tap + 1 < TAPS)
      copy_tap_A(wfrag + (tap + 1) * TAPFRAG, lds_a[(tap + 1) & 1], tid);
    __syncthreads();

    for (int idx = tid; idx < NTILE * NF; idx += 128) {
      int n = idx & 63;
      int c = idx >> 6;
      lds_s[n * NF + c] = s_val[n] ? x[c * THW + s_addr[n]] : (_Float16)0.f;
    }
    if (tap + 1 < TAPS) WAIT_ASYNC(4); else WAIT_ASYNC(0);
    __syncthreads();

    const _Float16* abuf = lds_a[tap & 1];
#pragma unroll
    for (int ks = 0; ks < KSTEPS; ++ks) {
      v16h a = *(const v16h*)(abuf + ((ks * MBLK + wid) * 32 + lane) * 16);
#pragma unroll
      for (int nb = 0; nb < 4; ++nb) {
        v16h b = *(const v16h*)(lds_s + (nb * 16 + nl) * NF + ks * 32 + g * 16);
        acc[nb] = __builtin_amdgcn_wmma_f32_16x16x32_f16(
            false, a, false, b, (short)0, acc[nb], false, false);
      }
    }
    __syncthreads();
  }

  for (int nb = 0; nb < 4; ++nb) {
    int n = base + nb * 16 + nl;
#pragma unroll
    for (int j = 0; j < 8; ++j) {
      int o = wid * 16 + g * 8 + j;
      if (o < OFFC) out[o * NPOS + n] = acc[nb][j] + bias[o];
    }
  }
}

// ---------------------------------------------------------------------------
// Deformable accumulation with WMMA. Sampling source is f16; offsets fp32.
// ---------------------------------------------------------------------------
__global__ __launch_bounds__(128)
void dcn_main_kernel(const _Float16* __restrict__ xs,
                     const float* __restrict__ off,
                     const _Float16* __restrict__ wfrag,
                     const float* __restrict__ bias,
                     const float* __restrict__ resid,   // nullptr or f32
                     float* __restrict__ outf,          // nullptr or f32 out
                     _Float16* __restrict__ outh,       // nullptr or f16 out
                     int leaky) {
  __shared__ alignas(32) _Float16 lds_s[NTILE * NF];
  __shared__ alignas(16) _Float16 lds_a[2][TAPFRAG];
  __shared__ int   s_a00[NTILE], s_a01[NTILE], s_a10[NTILE], s_a11[NTILE];
  __shared__ float s_w00[NTILE], s_w01[NTILE], s_w10[NTILE], s_w11[NTILE];

  const int tid  = threadIdx.x;
  const int lane = tid & 31;
  const int wid  = tid >> 5;
  const int g    = lane >> 4;
  const int nl   = lane & 15;
  const int base = blockIdx.x * NTILE;

  const v8f vzero = {0.f, 0.f, 0.f, 0.f, 0.f, 0.f, 0.f, 0.f};
  v8f acc[4];
  acc[0] = vzero; acc[1] = vzero; acc[2] = vzero; acc[3] = vzero;

  copy_tap_A(wfrag, lds_a[0], tid);

  for (int tap = 0; tap < TAPS; ++tap) {
    const int kt = tap / 9;
    const int kh = (tap / 3) % 3;
    const int kw = tap % 3;

    if (tid < NTILE) {
      int pos = base + tid;
      int t = pos / HW;  int r = pos - t * HW;
      int h = r / WW;    int w = r - h * WW;

      int ts = t + kt - 1;
      float tval = (ts >= 0 && ts < TT) ? 1.f : 0.f;
      ts = ts < 0 ? 0 : (ts >= TT ? TT - 1 : ts);

      float ph = (float)h + (float)(kh - 1) + off[(2 * tap)     * NPOS + pos];
      float pw = (float)w + (float)(kw - 1) + off[(2 * tap + 1) * NPOS + pos];
      float h0f = floorf(ph), w0f = floorf(pw);
      float lh = ph - h0f,    lw = pw - w0f;
      int h0 = (int)h0f, w0 = (int)w0f;

      float vh0 = (h0     >= 0 && h0     < HH) ? 1.f : 0.f;
      float vh1 = (h0 + 1 >= 0 && h0 + 1 < HH) ? 1.f : 0.f;
      float vw0 = (w0     >= 0 && w0     < WW) ? 1.f : 0.f;
      float vw1 = (w0 + 1 >= 0 && w0 + 1 < WW) ? 1.f : 0.f;
      int h0c = h0 < 0 ? 0 : (h0 > HH - 1 ? HH - 1 : h0);
      int h1c = h0 + 1 < 0 ? 0 : (h0 + 1 > HH - 1 ? HH - 1 : h0 + 1);
      int w0c = w0 < 0 ? 0 : (w0 > WW - 1 ? WW - 1 : w0);
      int w1c = w0 + 1 < 0 ? 0 : (w0 + 1 > WW - 1 ? WW - 1 : w0 + 1);

      int tb = ts * HW;
      s_a00[tid] = tb + h0c * WW + w0c;
      s_a01[tid] = tb + h0c * WW + w1c;
      s_a10[tid] = tb + h1c * WW + w0c;
      s_a11[tid] = tb + h1c * WW + w1c;
      s_w00[tid] = tval * vh0 * vw0 * (1.f - lh) * (1.f - lw);
      s_w01[tid] = tval * vh0 * vw1 * (1.f - lh) * lw;
      s_w10[tid] = tval * vh1 * vw0 * lh * (1.f - lw);
      s_w11[tid] = tval * vh1 * vw1 * lh * lw;
    }
    if (tap + 1 < TAPS)
      copy_tap_A(wfrag + (tap + 1) * TAPFRAG, lds_a[(tap + 1) & 1], tid);
    __syncthreads();

    for (int idx = tid; idx < NTILE * NF; idx += 128) {
      int n = idx & 63;
      int c = idx >> 6;
      const _Float16* xc = xs + c * THW;
      float v = s_w00[n] * (float)xc[s_a00[n]] + s_w01[n] * (float)xc[s_a01[n]] +
                s_w10[n] * (float)xc[s_a10[n]] + s_w11[n] * (float)xc[s_a11[n]];
      lds_s[n * NF + c] = (_Float16)v;
    }
    if (tap + 1 < TAPS) WAIT_ASYNC(4); else WAIT_ASYNC(0);
    __syncthreads();

    const _Float16* abuf = lds_a[tap & 1];
#pragma unroll
    for (int ks = 0; ks < KSTEPS; ++ks) {
      v16h a = *(const v16h*)(abuf + ((ks * MBLK + wid) * 32 + lane) * 16);
#pragma unroll
      for (int nb = 0; nb < 4; ++nb) {
        v16h b = *(const v16h*)(lds_s + (nb * 16 + nl) * NF + ks * 32 + g * 16);
        acc[nb] = __builtin_amdgcn_wmma_f32_16x16x32_f16(
            false, a, false, b, (short)0, acc[nb], false, false);
      }
    }
    __syncthreads();
  }

  for (int nb = 0; nb < 4; ++nb) {
    int n = base + nb * 16 + nl;
#pragma unroll
    for (int j = 0; j < 8; ++j) {
      int o = wid * 16 + g * 8 + j;
      float v = acc[nb][j] + bias[o];
      if (leaky) v = v > 0.f ? v : 0.1f * v;
      if (resid) v += resid[o * NPOS + n];
      if (outf)  outf[o * NPOS + n] = v;
      if (outh)  outh[o * NPOS + n] = (_Float16)v;
    }
  }
}

// ---------------------------------------------------------------------------
extern "C" void kernel_launch(void* const* d_in, const int* in_sizes, int n_in,
                              void* d_out, int out_size, void* d_ws, size_t ws_size,
                              hipStream_t stream) {
  (void)in_sizes; (void)n_in; (void)out_size; (void)ws_size;

  const float* x     = (const float*)d_in[0];
  const float* woff0 = (const float*)d_in[1];
  const float* boff0 = (const float*)d_in[2];
  const float* w0    = (const float*)d_in[3];
  const float* b0    = (const float*)d_in[4];
  const float* woff1 = (const float*)d_in[5];
  const float* boff1 = (const float*)d_in[6];
  const float* w1    = (const float*)d_in[7];
  const float* b1    = (const float*)d_in[8];
  float* out = (float*)d_out;

  char* wsb = (char*)d_ws;
  size_t cur = 0;
  auto carve = [&](size_t bytes) -> void* {
    void* p = (void*)(wsb + cur);
    cur += (bytes + 255) & ~(size_t)255;
    return p;
  };

  float*    offbuf = (float*)   carve((size_t)OFFC * NPOS * sizeof(float));
  _Float16* xh     = (_Float16*)carve((size_t)NF * NPOS * sizeof(_Float16));
  _Float16* yh     = (_Float16*)carve((size_t)NF * NPOS * sizeof(_Float16));
  _Float16* fwoff0 = (_Float16*)carve((size_t)FRAGTOTAL * sizeof(_Float16));
  _Float16* fw0    = (_Float16*)carve((size_t)FRAGTOTAL * sizeof(_Float16));
  _Float16* fwoff1 = (_Float16*)carve((size_t)FRAGTOTAL * sizeof(_Float16));
  _Float16* fw1    = (_Float16*)carve((size_t)FRAGTOTAL * sizeof(_Float16));

  const int nx = NF * NPOS;
  cvt_f32_to_f16_kernel<<<(nx + 255) / 256, 256, 0, stream>>>(x, xh, nx);

  const int pb = (FRAGTOTAL + 255) / 256;
  pack_wfrag_kernel<<<pb, 256, 0, stream>>>(woff0, OFFC, fwoff0);
  pack_wfrag_kernel<<<pb, 256, 0, stream>>>(w0,    NF,   fw0);
  pack_wfrag_kernel<<<pb, 256, 0, stream>>>(woff1, OFFC, fwoff1);
  pack_wfrag_kernel<<<pb, 256, 0, stream>>>(w1,    NF,   fw1);

  // Layer 1: y = leaky(DCN(x))  (y kept f16 for layer-2 sampling/GEMM)
  offset_conv_kernel<<<NTILES, 128, 0, stream>>>(xh, fwoff0, boff0, offbuf);
  dcn_main_kernel<<<NTILES, 128, 0, stream>>>(xh, offbuf, fw0, b0,
                                              nullptr, nullptr, yh, 1);
  // Layer 2: out = DCN(y) + x
  offset_conv_kernel<<<NTILES, 128, 0, stream>>>(yh, fwoff1, boff1, offbuf);
  dcn_main_kernel<<<NTILES, 128, 0, stream>>>(yh, offbuf, fw1, b1,
                                              x, out, nullptr, 0);
}
